// MoEnhanceTaskBlock_18528488915484
// MI455X (gfx1250) — matrix operand order repaired and checked by reading, not compiled, not run on hardware
//
#include <hip/hip_runtime.h>
#include <hip/hip_bf16.h>
#include <math.h>

typedef __attribute__((ext_vector_type(16))) _Float16 v16h;
typedef __attribute__((ext_vector_type(8)))  _Float16 v8h;
typedef __attribute__((ext_vector_type(8)))  float    v8f;

#define TOK   4096
#define CDIM  512
#define HEADS 8
#define HD    64
#define EATT  24
#define EFFD  16
#define KFFD  2
#define HID   1024
#define NSEQ  512
#define NB    8
#define CAP   8448   // 8192 assignments + 16*16 padding

// ---------------------------------------------------------------------------
// Wave-level WMMA helper: one wave computes a 16(M) x 64(N) f32 tile.
// A: per-lane row pointer (row-major, f16). B: BT[n][k] row-major f16.
// Layouts follow CDNA5 ISA 7.12.2 (16-bit A 16x32, B 32x16, 32-bit C 16x16).
// ---------------------------------------------------------------------------
__device__ __forceinline__ void wave_mma(v8f c[4],
                                         const _Float16* __restrict__ arow,
                                         const _Float16* __restrict__ BT,
                                         int ldb, int n0, int K, int lane) {
  const int koffA = (lane >> 4) << 3;   // lanes 16-31 hold K+8..15 / K+24..31
  const int koffB = (lane >> 4) << 3;   // lanes 16-31 hold K+8..15 / K+24..31
  const int nlane = lane & 15;
  for (int k = 0; k < K; k += 32) {
    v8h lo = *(const v8h*)(arow + k + koffA);
    v8h hi = *(const v8h*)(arow + k + koffA + 16);
    v16h a;
#pragma unroll
    for (int i = 0; i < 8; ++i) { a[i] = lo[i]; a[i + 8] = hi[i]; }
#pragma unroll
    for (int j = 0; j < 4; ++j) {
      const _Float16* brow = BT + (size_t)(n0 + j * 16 + nlane) * ldb + k + koffB;
      v8h blo = *(const v8h*)brow;
      v8h bhi = *(const v8h*)(brow + 16);
      v16h b;
#pragma unroll
      for (int i = 0; i < 8; ++i) { b[i] = blo[i]; b[i + 8] = bhi[i]; }
      c[j] = __builtin_amdgcn_wmma_f32_16x16x32_f16(false, a, false, b,
                                                    (short)0, c[j], false, false);
    }
  }
}

__device__ __forceinline__ void zero_c(v8f c[4]) {
#pragma unroll
  for (int j = 0; j < 4; ++j)
#pragma unroll
    for (int i = 0; i < 8; ++i) c[j][i] = 0.0f;
}

__device__ __forceinline__ void store_f32(const v8f c[4], float* C, int ldc,
                                          int m0, int n0, int lane, float alpha) {
  int r0 = m0 + ((lane >> 4) << 3);
  int c0 = n0 + (lane & 15);
#pragma unroll
  for (int j = 0; j < 4; ++j)
#pragma unroll
    for (int i = 0; i < 8; ++i)
      C[(size_t)(r0 + i) * ldc + c0 + j * 16] = c[j][i] * alpha;
}

__device__ __forceinline__ void store_f16(const v8f c[4], _Float16* C, int ldc,
                                          int m0, int n0, int lane) {
  int r0 = m0 + ((lane >> 4) << 3);
  int c0 = n0 + (lane & 15);
#pragma unroll
  for (int j = 0; j < 4; ++j)
#pragma unroll
    for (int i = 0; i < 8; ++i)
      C[(size_t)(r0 + i) * ldc + c0 + j * 16] = (_Float16)c[j][i];
}

// ---------------------------------------------------------------------------
// LayerNorm: one wave per token, 512 channels.
// ---------------------------------------------------------------------------
__global__ void k_ln(const float* __restrict__ x, const float* __restrict__ g,
                     const float* __restrict__ bb, _Float16* __restrict__ out) {
  int lane = threadIdx.x & 31;
  int t = blockIdx.x * 8 + (threadIdx.x >> 5);
  const float* xr = x + (size_t)t * CDIM;
  float v[16], s = 0.f;
#pragma unroll
  for (int i = 0; i < 16; ++i) { v[i] = xr[lane + 32 * i]; s += v[i]; }
  for (int o = 16; o; o >>= 1) s += __shfl_xor(s, o, 32);
  float mean = s * (1.f / CDIM);
  float q = 0.f;
#pragma unroll
  for (int i = 0; i < 16; ++i) { float d = v[i] - mean; q += d * d; }
  for (int o = 16; o; o >>= 1) q += __shfl_xor(q, o, 32);
  float inv = rsqrtf(q * (1.f / CDIM) + 1e-5f);
#pragma unroll
  for (int i = 0; i < 16; ++i) {
    int cc = lane + 32 * i;
    out[(size_t)t * CDIM + cc] = (_Float16)((v[i] - mean) * inv * g[cc] + bb[cc]);
  }
}

// ---------------------------------------------------------------------------
// Gating (attention experts): one thread per token.
// ---------------------------------------------------------------------------
__global__ void k_gate_att(const _Float16* __restrict__ x1h,
                           const float* __restrict__ gate_att,
                           const int* __restrict__ task,
                           int* __restrict__ ids, float* __restrict__ gates,
                           float* __restrict__ psum, float* __restrict__ freqs,
                           float* __restrict__ zacc) {
  int t = blockIdx.x * blockDim.x + threadIdx.x;
  if (t >= TOK) return;
  const float* w = gate_att + (size_t)task[0] * CDIM * EATT;
  float lg[EATT];
#pragma unroll
  for (int e = 0; e < EATT; ++e) lg[e] = 0.f;
  for (int cc = 0; cc < CDIM; ++cc) {
    float xv = (float)x1h[(size_t)t * CDIM + cc];
    const float* wr = w + cc * EATT;
    for (int e = 0; e < EATT; ++e) lg[e] += xv * wr[e];
  }
  float mx = lg[0];
  for (int e = 1; e < EATT; ++e) mx = fmaxf(mx, lg[e]);
  float se = 0.f, p[EATT];
  for (int e = 0; e < EATT; ++e) { p[e] = __expf(lg[e] - mx); se += p[e]; }
  float inv = 1.f / se;
  for (int e = 0; e < EATT; ++e) { p[e] *= inv; atomicAdd(&psum[e], p[e]); }
  float lse = mx + logf(se);
  atomicAdd(zacc, lse * lse);
  float q[EATT];
  for (int e = 0; e < EATT; ++e) q[e] = p[e];
  int sel[HEADS]; float val[HEADS]; float vsum = 0.f;
  for (int k = 0; k < HEADS; ++k) {
    int am = 0; float bv = -1.f;
    for (int e = 0; e < EATT; ++e) if (q[e] > bv) { bv = q[e]; am = e; }
    sel[k] = am; val[k] = bv; vsum += bv; q[am] = -1.f;
    atomicAdd(&freqs[am], 1.f);
  }
  float dn = 1.f / (vsum + 1e-6f);
  for (int k = 0; k < HEADS; ++k) {
    ids[t * HEADS + k] = sel[k];
    gates[t * HEADS + k] = val[k] * dn;
  }
}

// ---------------------------------------------------------------------------
// Gating (MLP experts): one thread per token, top-2 of 16, count per expert.
// ---------------------------------------------------------------------------
__global__ void k_gate_mlp(const _Float16* __restrict__ x2h,
                           const float* __restrict__ gate_mlp,
                           const int* __restrict__ task,
                           int* __restrict__ ids, float* __restrict__ gates,
                           float* __restrict__ psum, float* __restrict__ freqs,
                           float* __restrict__ zacc, int* __restrict__ cnt) {
  int t = blockIdx.x * blockDim.x + threadIdx.x;
  if (t >= TOK) return;
  const float* w = gate_mlp + (size_t)task[0] * CDIM * EFFD;
  float lg[EFFD];
#pragma unroll
  for (int e = 0; e < EFFD; ++e) lg[e] = 0.f;
  for (int cc = 0; cc < CDIM; ++cc) {
    float xv = (float)x2h[(size_t)t * CDIM + cc];
    const float* wr = w + cc * EFFD;
    for (int e = 0; e < EFFD; ++e) lg[e] += xv * wr[e];
  }
  float mx = lg[0];
  for (int e = 1; e < EFFD; ++e) mx = fmaxf(mx, lg[e]);
  float se = 0.f, p[EFFD];
  for (int e = 0; e < EFFD; ++e) { p[e] = __expf(lg[e] - mx); se += p[e]; }
  float inv = 1.f / se;
  for (int e = 0; e < EFFD; ++e) { p[e] *= inv; atomicAdd(&psum[e], p[e]); }
  float lse = mx + logf(se);
  atomicAdd(zacc, lse * lse);
  float q[EFFD];
  for (int e = 0; e < EFFD; ++e) q[e] = p[e];
  int sel[KFFD]; float val[KFFD]; float vsum = 0.f;
  for (int k = 0; k < KFFD; ++k) {
    int am = 0; float bv = -1.f;
    for (int e = 0; e < EFFD; ++e) if (q[e] > bv) { bv = q[e]; am = e; }
    sel[k] = am; val[k] = bv; vsum += bv; q[am] = -1.f;
    atomicAdd(&freqs[am], 1.f);
    atomicAdd(&cnt[am], 1);
  }
  float dn = 1.f / (vsum + 1e-6f);
  for (int k = 0; k < KFFD; ++k) {
    ids[t * KFFD + k] = sel[k];
    gates[t * KFFD + k] = val[k] * dn;
  }
}

// ---------------------------------------------------------------------------
// Weight pack/transpose kernels (fp32 -> f16, BT[n][k] layouts)
// ---------------------------------------------------------------------------
__global__ void k_pack_wqT(const float* __restrict__ wq, _Float16* __restrict__ d) {
  size_t i = (size_t)blockIdx.x * blockDim.x + threadIdx.x;          // n*512+c
  if (i >= (size_t)1536 * 512) return;
  int n = (int)(i / 512), cc = (int)(i % 512);
  d[i] = (_Float16)wq[(size_t)(n >> 6) * 512 * 64 + (size_t)cc * 64 + (n & 63)];
}
__global__ void k_pack_kvT(const float* __restrict__ kw, _Float16* __restrict__ d) {
  size_t i = (size_t)blockIdx.x * blockDim.x + threadIdx.x;          // n*512+c
  if (i >= (size_t)128 * 512) return;
  int n = (int)(i / 512), cc = (int)(i % 512);
  d[i] = (_Float16)kw[(size_t)cc * 128 + n];
}
__global__ void k_pack_woT(const float* __restrict__ wo, _Float16* __restrict__ d) {
  size_t i = (size_t)blockIdx.x * blockDim.x + threadIdx.x;          // c*1536+k
  if (i >= (size_t)512 * 1536) return;
  int cc = (int)(i / 1536), k = (int)(i % 1536);
  d[i] = (_Float16)wo[(size_t)k * 512 + cc];
}
__global__ void k_pack_wiT(const float* __restrict__ wi, _Float16* __restrict__ d) {
  size_t i = (size_t)blockIdx.x * blockDim.x + threadIdx.x;          // e,n,c
  if (i >= (size_t)EFFD * HID * CDIM) return;
  int e = (int)(i / ((size_t)HID * CDIM));
  int r = (int)(i % ((size_t)HID * CDIM));
  int n = r / CDIM, cc = r % CDIM;
  d[i] = (_Float16)wi[(size_t)e * CDIM * HID + (size_t)cc * HID + n];
}
__global__ void k_pack_wmT(const float* __restrict__ wm, _Float16* __restrict__ d) {
  size_t i = (size_t)blockIdx.x * blockDim.x + threadIdx.x;          // e,c,n
  if (i >= (size_t)EFFD * CDIM * HID) return;
  int e = (int)(i / ((size_t)CDIM * HID));
  int r = (int)(i % ((size_t)CDIM * HID));
  int cc = r / HID, n = r % HID;
  d[i] = (_Float16)wm[(size_t)e * HID * CDIM + (size_t)n * CDIM + cc];
}

// ---------------------------------------------------------------------------
// Dense GEMM kernels (block = 8 waves, tile 128x64)
// ---------------------------------------------------------------------------
__global__ void k_gemm_qall(const _Float16* __restrict__ A,
                            const _Float16* __restrict__ BT,
                            _Float16* __restrict__ C) {
  int lane = threadIdx.x & 31, wave = threadIdx.x >> 5;
  int m0 = blockIdx.x * 128 + wave * 16, n0 = blockIdx.y * 64;
  const _Float16* arow = A + (size_t)(m0 + (lane & 15)) * CDIM;
  v8f c[4]; zero_c(c);
  wave_mma(c, arow, BT, CDIM, n0, CDIM, lane);
  store_f16(c, C, 1536, m0, n0, lane);
}

__global__ void k_gemm_kv(const _Float16* __restrict__ A,
                          const _Float16* __restrict__ BT,
                          float* __restrict__ C) {
  int lane = threadIdx.x & 31, wave = threadIdx.x >> 5;
  int m0 = blockIdx.x * 128 + wave * 16, n0 = blockIdx.y * 64;
  const _Float16* arow = A + (size_t)(m0 + (lane & 15)) * CDIM;
  v8f c[4]; zero_c(c);
  wave_mma(c, arow, BT, CDIM, n0, CDIM, lane);
  store_f32(c, C, 128, m0, n0, lane, 1.0f);
}

__global__ void k_gemm_scores(const _Float16* __restrict__ QH,
                              const _Float16* __restrict__ KH,
                              float* __restrict__ S, int b) {
  int lane = threadIdx.x & 31, wave = threadIdx.x >> 5;
  int h = blockIdx.z;
  int m0 = blockIdx.x * 128 + wave * 16, n0 = blockIdx.y * 64;
  const _Float16* A = QH + ((size_t)(b * HEADS + h) * NSEQ) * HD;
  const _Float16* BT = KH + (size_t)b * NSEQ * HD;   // k rows ARE BT[n][k]
  const _Float16* arow = A + (size_t)(m0 + (lane & 15)) * HD;
  v8f c[4]; zero_c(c);
  wave_mma(c, arow, BT, HD, n0, HD, lane);
  store_f32(c, S + (size_t)h * NSEQ * NSEQ, NSEQ, m0, n0, lane, 0.125f);
}

__global__ void k_softmax(const float* __restrict__ S, _Float16* __restrict__ P) {
  int lane = threadIdx.x & 31;
  int row = blockIdx.x * 8 + (threadIdx.x >> 5);   // 8*512 rows per batch
  const float* sr = S + (size_t)row * NSEQ;
  float v[16], mx = -3.0e38f;
#pragma unroll
  for (int i = 0; i < 16; ++i) { v[i] = sr[lane + 32 * i]; mx = fmaxf(mx, v[i]); }
  for (int o = 16; o; o >>= 1) mx = fmaxf(mx, __shfl_xor(mx, o, 32));
  float s = 0.f;
#pragma unroll
  for (int i = 0; i < 16; ++i) { v[i] = __expf(v[i] - mx); s += v[i]; }
  for (int o = 16; o; o >>= 1) s += __shfl_xor(s, o, 32);
  float inv = 1.f / s;
#pragma unroll
  for (int i = 0; i < 16; ++i)
    P[(size_t)row * NSEQ + lane + 32 * i] = (_Float16)(v[i] * inv);
}

__global__ void k_gemm_o(const _Float16* __restrict__ P,
                         const _Float16* __restrict__ VT,
                         float* __restrict__ O, int b) {
  int lane = threadIdx.x & 31, wave = threadIdx.x >> 5;
  int h = blockIdx.z;
  int m0 = blockIdx.x * 128 + wave * 16, n0 = blockIdx.y * 64;
  const _Float16* A = P + (size_t)h * NSEQ * NSEQ;
  const _Float16* BT = VT + (size_t)b * HD * NSEQ;
  const _Float16* arow = A + (size_t)(m0 + (lane & 15)) * NSEQ;
  v8f c[4]; zero_c(c);
  wave_mma(c, arow, BT, NSEQ, n0, NSEQ, lane);
  store_f32(c, O + ((size_t)(b * HEADS + h) * NSEQ) * HD, HD, m0, n0, lane, 1.0f);
}

__global__ void k_gemm_y(const _Float16* __restrict__ A,
                         const _Float16* __restrict__ BT,
                         float* __restrict__ C) {
  int lane = threadIdx.x & 31, wave = threadIdx.x >> 5;
  int m0 = blockIdx.x * 128 + wave * 16, n0 = blockIdx.y * 64;
  const _Float16* arow = A + (size_t)(m0 + (lane & 15)) * 1536;
  v8f c[4]; zero_c(c);
  wave_mma(c, arow, BT, 1536, n0, 1536, lane);
  store_f32(c, C, CDIM, m0, n0, lane, 1.0f);
}

// ---------------------------------------------------------------------------
// Small glue kernels
// ---------------------------------------------------------------------------
__global__ void k_kv_repack(const float* __restrict__ kv, const float* __restrict__ kvb,
                            _Float16* __restrict__ kh, _Float16* __restrict__ vT) {
  size_t i = (size_t)blockIdx.x * blockDim.x + threadIdx.x;  // t*128 + d
  if (i >= (size_t)TOK * 128) return;
  int t = (int)(i >> 7), d = (int)(i & 127);
  int b = t >> 9, n = t & 511;
  float v = kv[i] + kvb[d];
  if (d < 64) kh[((size_t)b * NSEQ + n) * HD + d] = (_Float16)v;
  else        vT[((size_t)b * HD + (d - 64)) * NSEQ + n] = (_Float16)v;
}

__global__ void k_gather_qh(const _Float16* __restrict__ qall,
                            const int* __restrict__ ids, _Float16* __restrict__ qh) {
  size_t i = (size_t)blockIdx.x * blockDim.x + threadIdx.x;  // t*512 + k*64 + d
  if (i >= (size_t)TOK * HEADS * HD) return;
  int t = (int)(i >> 9), r = (int)(i & 511);
  int k = r >> 6, d = r & 63;
  int e = ids[t * HEADS + k];
  int b = t >> 9, n = t & 511;
  qh[((size_t)(b * HEADS + k) * NSEQ + n) * HD + d] =
      qall[(size_t)t * 1536 + e * 64 + d];
}

__global__ void k_zero_f32(float* __restrict__ p, size_t n) {
  size_t i = (size_t)blockIdx.x * blockDim.x + threadIdx.x;
  if (i < n) p[i] = 0.f;
}

__global__ void k_scatter_comb(const float* __restrict__ O,
                               const int* __restrict__ ids,
                               const float* __restrict__ gates,
                               _Float16* __restrict__ comb) {
  size_t i = (size_t)blockIdx.x * blockDim.x + threadIdx.x;  // t*512 + k*64 + d
  if (i >= (size_t)TOK * HEADS * HD) return;
  int t = (int)(i >> 9), r = (int)(i & 511);
  int k = r >> 6, d = r & 63;
  int e = ids[t * HEADS + k];
  int b = t >> 9, n = t & 511;
  float v = gates[t * HEADS + k] * O[((size_t)(b * HEADS + k) * NSEQ + n) * HD + d];
  comb[(size_t)t * 1536 + e * 64 + d] = (_Float16)v;
}

__global__ void k_resid(const float* __restrict__ x, const float* __restrict__ y,
                        float* __restrict__ out) {
  size_t i = (size_t)blockIdx.x * blockDim.x + threadIdx.x;
  if (i < (size_t)TOK * CDIM) out[i] = x[i] + y[i];
}

__global__ void k_zero_small(float* __restrict__ fb, int* __restrict__ ib) {
  int i = threadIdx.x;
  if (i < 128) fb[i] = 0.f;
  if (i < 64) ib[i] = 0;
}

__global__ void k_offsets(int* __restrict__ ib) {
  // ib: [0..15] cnt, [16..32] base (17), [33..48] cursor
  if (threadIdx.x == 0) {
    int acc = 0;
    for (int e = 0; e < EFFD; ++e) {
      ib[16 + e] = acc;
      acc += (ib[e] + 15) & ~15;
    }
    ib[32] = acc;
    for (int e = 0; e < EFFD; ++e) ib[33 + e] = 0;
  }
}

__global__ void k_assign(const int* __restrict__ ids, int* __restrict__ ib,
                         int* __restrict__ rowlist, int* __restrict__ arow) {
  int t = blockIdx.x * blockDim.x + threadIdx.x;
  if (t >= TOK) return;
  for (int k = 0; k < KFFD; ++k) {
    int e = ids[t * KFFD + k];
    int pos = atomicAdd(&ib[33 + e], 1);
    int row = ib[16 + e] + pos;
    rowlist[row] = t;
    arow[t * KFFD + k] = row;
  }
}

// ---------------------------------------------------------------------------
// Expert GEMMs (gathered rows, ragged M per expert)
// ---------------------------------------------------------------------------
__global__ void k_gemm_h(const _Float16* __restrict__ X2H,
                         const _Float16* __restrict__ WIT,
                         const int* __restrict__ rowlist,
                         const int* __restrict__ ib,
                         const float* __restrict__ bi,
                         _Float16* __restrict__ H) {
  int lane = threadIdx.x & 31, wave = threadIdx.x >> 5;
  int e = blockIdx.z;
  int b0 = ib[16 + e];
  int padCnt = ib[16 + e + 1] - b0;
  int cnt = ib[e];
  int m0 = blockIdx.x * 128 + wave * 16;
  if (m0 >= padCnt) return;
  int mr = m0 + (lane & 15);
  int token = (mr < cnt) ? rowlist[b0 + mr] : 0;
  const _Float16* arow = X2H + (size_t)token * CDIM;
  int n0 = blockIdx.y * 64;
  v8f c[4]; zero_c(c);
  wave_mma(c, arow, WIT + (size_t)e * HID * CDIM, CDIM, n0, CDIM, lane);
  int r0 = b0 + m0 + ((lane >> 4) << 3);
  int c0 = n0 + (lane & 15);
#pragma unroll
  for (int j = 0; j < 4; ++j)
#pragma unroll
    for (int i = 0; i < 8; ++i) {
      float xv = c[j][i] + bi[e * HID + c0 + j * 16];
      float gl = 0.5f * xv * (1.f + erff(xv * 0.70710678f));
      H[(size_t)(r0 + i) * HID + c0 + j * 16] = (_Float16)gl;
    }
}

__global__ void k_gemm_ys(const _Float16* __restrict__ H,
                          const _Float16* __restrict__ WMT,
                          const int* __restrict__ ib,
                          float* __restrict__ YS) {
  int lane = threadIdx.x & 31, wave = threadIdx.x >> 5;
  int e = blockIdx.z;
  int b0 = ib[16 + e];
  int padCnt = ib[16 + e + 1] - b0;
  int m0 = blockIdx.x * 128 + wave * 16;
  if (m0 >= padCnt) return;
  const _Float16* arow = H + (size_t)(b0 + m0 + (lane & 15)) * HID;
  int n0 = blockIdx.y * 64;
  v8f c[4]; zero_c(c);
  wave_mma(c, arow, WMT + (size_t)e * CDIM * HID, HID, n0, HID, lane);
  store_f32(c, YS, CDIM, b0 + m0, n0, lane, 1.0f);
}

__global__ void k_combine(const float* __restrict__ g2, const int* __restrict__ ids2,
                          const int* __restrict__ arow, const float* __restrict__ YS,
                          const float* __restrict__ bo, float* __restrict__ xout) {
  size_t i = (size_t)blockIdx.x * blockDim.x + threadIdx.x;
  if (i >= (size_t)TOK * CDIM) return;
  int t = (int)(i >> 9), cc = (int)(i & 511);
  float acc = 0.f;
  for (int k = 0; k < KFFD; ++k) {
    float gk = g2[t * KFFD + k];
    int e = ids2[t * KFFD + k];
    int row = arow[t * KFFD + k];
    acc += gk * (YS[(size_t)row * CDIM + cc] + bo[e * CDIM + cc]);
  }
  xout[i] += acc;
}

__global__ void k_loss_final(const float* __restrict__ fb, float* __restrict__ out) {
  if (threadIdx.x != 0 || blockIdx.x != 0) return;
  float p1 = 0.f, f1 = 0.f;
  for (int e = 0; e < EATT; ++e) { p1 += fb[e]; f1 += fb[24 + e]; }
  float s1 = 0.f;
  for (int e = 0; e < EATT; ++e) s1 += (fb[e] / p1) * (fb[24 + e] / f1);
  s1 *= EATT;
  float z1 = fb[48] / (float)TOK;
  float p2 = 0.f, f2 = 0.f;
  for (int e = 0; e < EFFD; ++e) { p2 += fb[64 + e]; f2 += fb[80 + e]; }
  float s2 = 0.f;
  for (int e = 0; e < EFFD; ++e) s2 += (fb[64 + e] / p2) * (fb[80 + e] / f2);
  s2 *= EFFD;
  float z2 = fb[96] / (float)TOK;
  out[0] = 0.01f * (s1 + s2) + 0.001f * (z1 + z2);
}

// ---------------------------------------------------------------------------
extern "C" void kernel_launch(void* const* d_in, const int* in_sizes, int n_in,
                              void* d_out, int out_size, void* d_ws, size_t ws_size,
                              hipStream_t stream) {
  (void)in_sizes; (void)n_in; (void)out_size; (void)ws_size;
  const float* x       = (const float*)d_in[0];
  const int*   task    = (const int*)d_in[1];
  const float* n1g     = (const float*)d_in[2];
  const float* n1b     = (const float*)d_in[3];
  const float* gateA   = (const float*)d_in[4];
  const float* wq      = (const float*)d_in[5];
  const float* wo      = (const float*)d_in[6];
  const float* kvw     = (const float*)d_in[7];
  const float* kvb     = (const float*)d_in[8];
  const float* n2g     = (const float*)d_in[9];
  const float* n2b     = (const float*)d_in[10];
  const float* gateM   = (const float*)d_in[11];
  const float* wi      = (const float*)d_in[12];
  const float* bi      = (const float*)d_in[13];
  const float* wm      = (const float*)d_in[14];
  const float* bo      = (const float*)d_in[15];
  float* xout = (float*)d_out;                       // [TOK, CDIM] + loss at end
  float* lout = xout + (size_t)TOK * CDIM;

  char* ws = (char*)d_ws;
  size_t off = 0;
  auto alloc = [&](size_t bytes) { size_t o = off; off = (off + bytes + 255) & ~(size_t)255; return o; };
  size_t o_x1h  = alloc((size_t)TOK * CDIM * 2);            // reused as x2h
  size_t o_wqT  = alloc((size_t)1536 * 512 * 2);
  size_t o_kvT  = alloc((size_t)128 * 512 * 2);
  size_t o_woT  = alloc((size_t)512 * 1536 * 2);
  size_t o_wiT  = alloc((size_t)EFFD * HID * CDIM * 2);
  size_t o_wmT  = alloc((size_t)EFFD * CDIM * HID * 2);
  size_t o_qh   = alloc((size_t)NB * HEADS * NSEQ * HD * 2);
  size_t o_kh   = alloc((size_t)NB * NSEQ * HD * 2);
  size_t o_vT   = alloc((size_t)NB * HD * NSEQ * 2);
  size_t o_kvt  = alloc((size_t)TOK * 128 * 4);
  size_t o_ids1 = alloc((size_t)TOK * HEADS * 4);
  size_t o_g1   = alloc((size_t)TOK * HEADS * 4);
  size_t o_ids2 = alloc((size_t)TOK * KFFD * 4);
  size_t o_g2   = alloc((size_t)TOK * KFFD * 4);
  size_t o_ar   = alloc((size_t)TOK * KFFD * 4);
  size_t o_rl   = alloc((size_t)CAP * 4);
  size_t o_ib   = alloc(64 * 4);
  size_t o_fb   = alloc(128 * 4);
  // big phase-shared region: attention (qall/comb, S, P, O, Y) then MLP (H, YS)
  size_t o_qall = alloc((size_t)TOK * 1536 * 2);            // also comb
  size_t o_S    = alloc((size_t)HEADS * NSEQ * NSEQ * 4);
  size_t o_P    = alloc((size_t)HEADS * NSEQ * NSEQ * 2);
  size_t o_O    = alloc((size_t)NB * HEADS * NSEQ * HD * 4);
  size_t o_Y    = alloc((size_t)TOK * CDIM * 4);
  size_t o_H    = o_qall;                                   // CAP*HID*2   = 17.3MB
  size_t o_YS   = o_qall + ((size_t)CAP * HID * 2 + 255 & ~(size_t)255); // CAP*CDIM*4

  _Float16* x1h  = (_Float16*)(ws + o_x1h);
  _Float16* wqT  = (_Float16*)(ws + o_wqT);
  _Float16* kvT  = (_Float16*)(ws + o_kvT);
  _Float16* woT  = (_Float16*)(ws + o_woT);
  _Float16* wiT  = (_Float16*)(ws + o_wiT);
  _Float16* wmT  = (_Float16*)(ws + o_wmT);
  _Float16* qh   = (_Float16*)(ws + o_qh);
  _Float16* kh   = (_Float16*)(ws + o_kh);
  _Float16* vT   = (_Float16*)(ws + o_vT);
  float*    kvt  = (float*)(ws + o_kvt);
  int*      ids1 = (int*)(ws + o_ids1);
  float*    g1   = (float*)(ws + o_g1);
  int*      ids2 = (int*)(ws + o_ids2);
  float*    g2   = (float*)(ws + o_g2);
  int*      ar   = (int*)(ws + o_ar);
  int*      rl   = (int*)(ws + o_rl);
  int*      ib   = (int*)(ws + o_ib);
  float*    fb   = (float*)(ws + o_fb);
  _Float16* qall = (_Float16*)(ws + o_qall);
  float*    S    = (float*)(ws + o_S);
  _Float16* P    = (_Float16*)(ws + o_P);
  float*    O    = (float*)(ws + o_O);
  float*    Y    = (float*)(ws + o_Y);
  _Float16* H    = (_Float16*)(ws + o_H);
  float*    YS   = (float*)(ws + o_YS);

  // ---- init + pack weights ----
  k_zero_small<<<1, 256, 0, stream>>>(fb, ib);
  k_pack_wqT<<<(1536 * 512 + 255) / 256, 256, 0, stream>>>(wq, wqT);
  k_pack_kvT<<<(128 * 512 + 255) / 256, 256, 0, stream>>>(kvw, kvT);
  k_pack_woT<<<(512 * 1536 + 255) / 256, 256, 0, stream>>>(wo, woT);
  k_pack_wiT<<<(EFFD * HID * CDIM + 255) / 256, 256, 0, stream>>>(wi, wiT);
  k_pack_wmT<<<(EFFD * CDIM * HID + 255) / 256, 256, 0, stream>>>(wm, wmT);

  // ---- attention phase ----
  k_ln<<<TOK / 8, 256, 0, stream>>>(x, n1g, n1b, x1h);
  k_gate_att<<<TOK / 256, 256, 0, stream>>>(x1h, gateA, task, ids1, g1,
                                            fb, fb + 24, fb + 48);
  k_gemm_qall<<<dim3(TOK / 128, 1536 / 64), 256, 0, stream>>>(x1h, wqT, qall);
  k_gemm_kv<<<dim3(TOK / 128, 128 / 64), 256, 0, stream>>>(x1h, kvT, kvt);
  k_kv_repack<<<(TOK * 128 + 255) / 256, 256, 0, stream>>>(kvt, kvb, kh, vT);
  k_gather_qh<<<(TOK * HEADS * HD + 255) / 256, 256, 0, stream>>>(qall, ids1, qh);
  for (int b = 0; b < NB; ++b) {
    k_gemm_scores<<<dim3(NSEQ / 128, NSEQ / 64, HEADS), 256, 0, stream>>>(qh, kh, S, b);
    k_softmax<<<HEADS * NSEQ / 8, 256, 0, stream>>>(S, P);
    k_gemm_o<<<dim3(NSEQ / 128, 1, HEADS), 256, 0, stream>>>(P, vT, O, b);
  }
  // comb (aliases qall; qall fully consumed by gather above)
  k_zero_f32<<<(int)(((size_t)TOK * 1536 / 2 + 255) / 256), 256, 0, stream>>>(
      (float*)qall, (size_t)TOK * 1536 / 2);
  k_scatter_comb<<<(TOK * HEADS * HD + 255) / 256, 256, 0, stream>>>(O, ids1, g1, qall);
  k_gemm_y<<<dim3(TOK / 128, CDIM / 64), 256, 0, stream>>>(qall, woT, Y);
  k_resid<<<(TOK * CDIM + 255) / 256, 256, 0, stream>>>(x, Y, xout);

  // ---- MLP phase ----
  _Float16* x2h = x1h;  // reuse
  k_ln<<<TOK / 8, 256, 0, stream>>>(xout, n2g, n2b, x2h);
  k_gate_mlp<<<TOK / 256, 256, 0, stream>>>(x2h, gateM, task, ids2, g2,
                                            fb + 64, fb + 80, fb + 96, ib);
  k_offsets<<<1, 32, 0, stream>>>(ib);
  k_assign<<<TOK / 256, 256, 0, stream>>>(ids2, ib, rl, ar);
  k_gemm_h<<<dim3(CAP / 128, HID / 64, EFFD), 256, 0, stream>>>(x2h, wiT, rl, ib, bi, H);
  k_gemm_ys<<<dim3(CAP / 128, CDIM / 64, EFFD), 256, 0, stream>>>(H, wmT, ib, YS);
  k_combine<<<(TOK * CDIM + 255) / 256, 256, 0, stream>>>(g2, ids2, ar, YS, bo, xout);
  k_loss_final<<<1, 32, 0, stream>>>(fb, lout);
}
